// SemanticLSTM_83743272338028
// MI455X (gfx1250) — matrix operand order repaired and checked by reading, not compiled, not run on hardware
//
#include <hip/hip_runtime.h>
#include <hip/hip_bf16.h>
#include <math.h>

typedef __attribute__((ext_vector_type(16))) __bf16 v16bf;
typedef __attribute__((ext_vector_type(8)))  float  v8f;
typedef unsigned int u32x4 __attribute__((ext_vector_type(4)));

union FragB { v16bf v; u32x4 u[2]; };

// Global-address-space 16B load: guarantees global_load_b128 (LOADcnt only).
typedef __attribute__((address_space(1))) const u32x4 gc_u32x4;
__device__ __forceinline__ u32x4 gload16(const void* p) {
    return *(gc_u32x4*)p;
}

// ---------------------------------------------------------------------------
// Generic batched GEMM:  C[z] = A[z](MxK,bf16) * Wt[z](NxK,bf16)^T (+Cadd)(+bias)
// Requirements (caller guarantees by padding):
//   K % 64 == 0 ; all A rows [0, ceil64(M)) and W rows [0, 128*gridY) readable.
// Store is guarded by M/N. WG = 256 threads = 8 waves; wave tile 16Mx64N;
// WG tile 64M x 128N; K unrolled by 64 (2 chunks, 8 WMMAs per iteration).
// ---------------------------------------------------------------------------
__global__ __launch_bounds__(256) void k_gemm_bf16(
    const __bf16* __restrict__ A, long long sA, int lda,
    const __bf16* __restrict__ Wt, long long sW, int ldw,
    const float* __restrict__ Cadd, long long sCadd,
    const float* __restrict__ bias, long long sBias,
    void* __restrict__ Cout, long long sC, int ldc,
    int M, int N, int K, int outBf16, int remap, int RB, int RT)
{
    const int z = blockIdx.z;
    A  += (long long)z * sA;
    Wt += (long long)z * sW;

    const int tid  = threadIdx.x;
    const int wave = tid >> 5;
    const int lane = tid & 31;
    const int wm = wave & 3;        // 0..3 -> M sub-tile
    const int wn = wave >> 2;       // 0..1 -> N sub-tile
    const int m0 = blockIdx.x * 64  + wm * 16;
    const int n0 = blockIdx.y * 128 + wn * 64;
    const int l15 = lane & 15;
    const int lh  = lane >> 4;

    v8f acc[4];
#pragma unroll
    for (int j = 0; j < 4; ++j) {
#pragma unroll
        for (int v = 0; v < 8; ++v) acc[j][v] = 0.0f;
    }

    // A fragment: lane<16 -> row m0+l15, K {0..7, 16..23}; lane>=16 -> K {8..15, 24..31}
    const __bf16* __restrict__ aP = A + (long long)(m0 + l15) * lda + lh * 8;
    // B fragments: lane<16 -> col n, K {0..7, 16..23}; lane>=16 -> K {8..15, 24..31}
    const __bf16* __restrict__ bP[4];
#pragma unroll
    for (int j = 0; j < 4; ++j)
        bP[j] = Wt + (long long)(n0 + j * 16 + l15) * ldw + lh * 8;

    for (int k0 = 0; k0 < K; k0 += 64) {
        FragB aA, aB, bA[4], bB[4];
        // Issue all 20 b128 loads for the 64-deep double-chunk first.
        aA.u[0] = gload16(aP + k0);
        aA.u[1] = gload16(aP + k0 + 16);
        aB.u[0] = gload16(aP + k0 + 32);
        aB.u[1] = gload16(aP + k0 + 48);
#pragma unroll
        for (int j = 0; j < 4; ++j) {
            const __bf16* bp = bP[j] + k0;
            bA[j].u[0] = gload16(bp);
            bA[j].u[1] = gload16(bp + 16);
            bB[j].u[0] = gload16(bp + 32);
            bB[j].u[1] = gload16(bp + 48);
        }
        // 8 WMMAs into 4 independent accumulators.
#pragma unroll
        for (int j = 0; j < 4; ++j)
            acc[j] = __builtin_amdgcn_wmma_f32_16x16x32_bf16(
                false, aA.v, false, bA[j].v, (short)0, acc[j], false, false);
#pragma unroll
        for (int j = 0; j < 4; ++j)
            acc[j] = __builtin_amdgcn_wmma_f32_16x16x32_bf16(
                false, aB.v, false, bB[j].v, (short)0, acc[j], false, false);
    }

    float*  outF = (float*)Cout;
    __bf16* outH = (__bf16*)Cout;
#pragma unroll
    for (int j = 0; j < 4; ++j) {
        int n = n0 + j * 16 + l15;
        if (n >= N) continue;
#pragma unroll
        for (int v = 0; v < 8; ++v) {
            int m = m0 + v + (lh << 3);
            if (m >= M) continue;
            float val = acc[j][v];
            if (Cadd) val += Cadd[(long long)z * sCadd + (long long)m * ldc + n];
            if (bias) val += bias[(long long)z * sBias + n];
            long long orow = remap ? ((long long)(m % RB) * RT + (m / RB)) : (long long)m;
            long long idx  = (long long)z * sC + orow * (long long)ldc + n;
            if (outBf16) outH[idx] = (__bf16)val;
            else         outF[idx] = val;
        }
    }
}

// fp32 -> bf16 with zero-padding: source is rs x cs, dest is (total/cd) x cd.
__global__ __launch_bounds__(256) void k_convert_pad(
    const float* __restrict__ src, __bf16* __restrict__ dst,
    int total, int rs, int cs, int cd)
{
    int i = blockIdx.x * 256 + threadIdx.x;
    if (i >= total) return;
    int r = i / cd, c = i % cd;
    dst[i] = (r < rs && c < cs) ? (__bf16)src[(long long)r * cs + c] : (__bf16)0.0f;
}

// we (T,B,E) bf16: t==0 -> feat2input(feats); t>0 -> word_embed[seq[b,t-1]]
__global__ __launch_bounds__(256) void k_build_we(
    const float* __restrict__ we0, const float* __restrict__ wemb,
    const int* __restrict__ seq, __bf16* __restrict__ we)
{
    int i = blockIdx.x * 256 + threadIdx.x;
    if (i >= 3200 * 512) return;
    int r = i >> 9, e = i & 511;
    int t = r >> 7, b = r & 127;
    float v;
    if (t == 0) v = we0[(b << 9) + e];
    else        v = wemb[((long long)seq[b * 24 + (t - 1)] << 9) + e];
    we[i] = (__bf16)v;
}

// second[g,b,h] = tmp3[b,g*512+h] * tmp4[b,g*512+h]  (bf16)
__global__ __launch_bounds__(256) void k_prep_second(
    const float* __restrict__ t3, const float* __restrict__ t4,
    __bf16* __restrict__ out)
{
    int i = blockIdx.x * 256 + threadIdx.x;
    if (i >= 4 * 128 * 512) return;
    int g = i >> 16, rem = i & 65535;
    int b = rem >> 9, h = rem & 511;
    int col = (g << 9) + h;
    out[i] = (__bf16)(t3[(b << 11) + col] * t4[(b << 11) + col]);
}

// a[g,b,h] = tmp1[t*128+b, g*512+h] * tmp2[b, g*512+h]  (bf16)
__global__ __launch_bounds__(256) void k_prep_at(
    const __bf16* __restrict__ tmp1, const float* __restrict__ tmp2,
    __bf16* __restrict__ out, int t)
{
    int i = blockIdx.x * 256 + threadIdx.x;
    if (i >= 4 * 128 * 512) return;
    int g = i >> 16, rem = i & 65535;
    int b = rem >> 9, h = rem & 511;
    int col = (g << 9) + h;
    float v = (float)tmp1[(((long long)(t * 128 + b)) << 11) + col] * tmp2[(b << 11) + col];
    out[i] = (__bf16)v;
}

// hb[g,b,h] = hid[g,b,h] * tag_gate[b, g*512+h]  (bf16)
__global__ __launch_bounds__(256) void k_prep_hb(
    const float* __restrict__ hid, const float* __restrict__ tg,
    __bf16* __restrict__ out)
{
    int i = blockIdx.x * 256 + threadIdx.x;
    if (i >= 4 * 128 * 512) return;
    int g = i >> 16, rem = i & 65535;
    int b = rem >> 9, h = rem & 511;
    out[i] = (__bf16)(hid[i] * tg[(b << 11) + ((g << 9) + h)]);
}

// LSTM pointwise: gates from pre(4,B,H); update c (fp32), h (bf16), hs[t] (bf16)
__global__ __launch_bounds__(256) void k_lstm(
    const float* __restrict__ pre, float* __restrict__ c,
    __bf16* __restrict__ h, __bf16* __restrict__ hs, int t)
{
    int i = blockIdx.x * 256 + threadIdx.x;
    if (i >= 128 * 512) return;
    float ig = 1.0f / (1.0f + expf(-pre[i]));
    float fg = 1.0f / (1.0f + expf(-pre[65536 + i]));
    float og = 1.0f / (1.0f + expf(-pre[131072 + i]));
    float gg = tanhf(pre[196608 + i]);
    float cn = fg * c[i] + ig * gg;
    float hn = og * tanhf(cn);
    c[i] = cn;
    h[i] = (__bf16)hn;
    hs[((long long)t << 16) + i] = (__bf16)hn;
}

// in-place log_softmax per row (ncols), one 256-thread WG per row
__global__ __launch_bounds__(256) void k_logsoftmax(float* __restrict__ out, int ncols)
{
    __shared__ float red[256];
    const long long base = (long long)blockIdx.x * ncols;
    const int tid = threadIdx.x;

    float m = -INFINITY;
    for (int c = tid; c < ncols; c += 256) m = fmaxf(m, out[base + c]);
    red[tid] = m; __syncthreads();
    for (int s = 128; s > 0; s >>= 1) {
        if (tid < s) red[tid] = fmaxf(red[tid], red[tid + s]);
        __syncthreads();
    }
    m = red[0]; __syncthreads();

    float sum = 0.0f;
    for (int c = tid; c < ncols; c += 256) sum += expf(out[base + c] - m);
    red[tid] = sum; __syncthreads();
    for (int s = 128; s > 0; s >>= 1) {
        if (tid < s) red[tid] += red[tid + s];
        __syncthreads();
    }
    float lse = m + logf(red[0]);

    for (int c = tid; c < ncols; c += 256) out[base + c] -= lse;
}

// ---------------------------------------------------------------------------
static inline int nblk(long long n) { return (int)((n + 255) / 256); }

static void gemm(hipStream_t s,
                 const __bf16* A, long long sA, int lda,
                 const __bf16* W, long long sW, int ldw,
                 const float* Cadd, long long sCadd,
                 const float* bias, long long sBias,
                 void* C, long long sC, int ldc,
                 int M, int N, int K, int Z,
                 int outBf16, int remap, int RB, int RT)
{
    dim3 g((M + 63) / 64, (N + 127) / 128, Z);
    k_gemm_bf16<<<g, 256, 0, s>>>(A, sA, lda, W, sW, ldw, Cadd, sCadd,
                                  bias, sBias, C, sC, ldc,
                                  M, N, K, outBf16, remap, RB, RT);
}

extern "C" void kernel_launch(void* const* d_in, const int* in_sizes, int n_in,
                              void* d_out, int out_size, void* d_ws, size_t ws_size,
                              hipStream_t stream)
{
    (void)in_sizes; (void)n_in; (void)out_size; (void)ws_size;

    const float* feats = (const float*)d_in[0];
    const float* tags  = (const float*)d_in[1];
    const int*   seq   = (const int*)d_in[2];
    const float* wemb  = (const float*)d_in[3];
    const float* Wf2i  = (const float*)d_in[4];
    const float* Wt1   = (const float*)d_in[5];
    const float* Wf2l  = (const float*)d_in[6];
    const float* Wt2   = (const float*)d_in[7];
    const float* Ww2l  = (const float*)d_in[8];
    const float* Wgate = (const float*)d_in[9];
    const float* bgate = (const float*)d_in[10];
    const float* Whid  = (const float*)d_in[11];
    const float* Wtg   = (const float*)d_in[12];
    const float* Wboth = (const float*)d_in[13];
    const float* Ww2lo = (const float*)d_in[14];
    const float* lbias = (const float*)d_in[15];

    char* base = (char*)d_ws;
    size_t off = 0;
    auto alloc = [&](size_t bytes) -> void* {
        void* p = base + off;
        off = (off + bytes + 255) & ~(size_t)255;
        return p;
    };

    const int VP = 12032;   // V-2 (=11998) padded to 94 * 128 so all B loads are in-bounds

    // bf16 operands (K padded to mult of 32; V-2 padded to VP rows)
    __bf16* tags_bf  = (__bf16*)alloc((size_t)128 * 320 * 2);
    __bf16* feats_bf = (__bf16*)alloc((size_t)128 * 2048 * 2);
    __bf16* Wt1_bf   = (__bf16*)alloc((size_t)2048 * 320 * 2);
    __bf16* Wt2_bf   = (__bf16*)alloc((size_t)2048 * 320 * 2);
    __bf16* Wtg_bf   = (__bf16*)alloc((size_t)2048 * 320 * 2);
    __bf16* Wf2l_bf  = (__bf16*)alloc((size_t)2048 * 2048 * 2);
    __bf16* Wf2i_bf  = (__bf16*)alloc((size_t)512 * 2048 * 2);
    __bf16* Ww2l_bf  = (__bf16*)alloc((size_t)2048 * 512 * 2);
    __bf16* Wgate_bf = (__bf16*)alloc((size_t)2048 * 1024 * 2);
    __bf16* Whid_bf  = (__bf16*)alloc((size_t)2048 * 512 * 2);
    __bf16* Wboth_bf = (__bf16*)alloc((size_t)2048 * 512 * 2);
    __bf16* Ww2lo_bf = (__bf16*)alloc((size_t)512 * 512 * 2);
    __bf16* wemb_bf  = (__bf16*)alloc((size_t)VP * 512 * 2);
    __bf16* we_bf    = (__bf16*)alloc((size_t)3200 * 512 * 2);
    __bf16* tmp1_bf  = (__bf16*)alloc((size_t)3200 * 2048 * 2);
    __bf16* second_bf= (__bf16*)alloc((size_t)4 * 128 * 512 * 2);
    __bf16* a_bf     = (__bf16*)alloc((size_t)4 * 128 * 512 * 2);
    __bf16* hb_bf    = (__bf16*)alloc((size_t)4 * 128 * 512 * 2);
    __bf16* h_bf     = (__bf16*)alloc((size_t)128 * 512 * 2);
    __bf16* hs_bf    = (__bf16*)alloc((size_t)3200 * 512 * 2);
    __bf16* lwT_bf   = (__bf16*)alloc((size_t)VP * 512 * 2);
    // fp32 intermediates
    float* tmp2_f = (float*)alloc((size_t)128 * 2048 * 4);
    float* tmp3_f = (float*)alloc((size_t)128 * 2048 * 4);
    float* tmp4_f = (float*)alloc((size_t)128 * 2048 * 4);
    float* tg_f   = (float*)alloc((size_t)128 * 2048 * 4);
    float* we0_f  = (float*)alloc((size_t)128 * 512 * 4);
    float* inpc_f = (float*)alloc((size_t)4 * 128 * 512 * 4);
    float* inp_f  = (float*)alloc((size_t)4 * 128 * 512 * 4);
    float* hid_f  = (float*)alloc((size_t)4 * 128 * 512 * 4);
    float* pre_f  = (float*)alloc((size_t)4 * 128 * 512 * 4);
    float* c_f    = (float*)alloc((size_t)128 * 512 * 4);

    // -------- conversions (fp32 -> bf16, pad K=300->320, V-2 -> VP rows) -----
    k_convert_pad<<<nblk(128LL*320), 256, 0, stream>>>(tags,  tags_bf,  128*320,  128, 300, 320);
    k_convert_pad<<<nblk(128LL*2048),256, 0, stream>>>(feats, feats_bf, 128*2048, 128,2048,2048);
    k_convert_pad<<<nblk(2048LL*320),256, 0, stream>>>(Wt1,   Wt1_bf,   2048*320, 2048,300, 320);
    k_convert_pad<<<nblk(2048LL*320),256, 0, stream>>>(Wt2,   Wt2_bf,   2048*320, 2048,300, 320);
    k_convert_pad<<<nblk(2048LL*320),256, 0, stream>>>(Wtg,   Wtg_bf,   2048*320, 2048,300, 320);
    k_convert_pad<<<nblk(2048LL*2048),256,0, stream>>>(Wf2l,  Wf2l_bf,  2048*2048,2048,2048,2048);
    k_convert_pad<<<nblk(512LL*2048),256, 0, stream>>>(Wf2i,  Wf2i_bf,  512*2048, 512,2048,2048);
    k_convert_pad<<<nblk(2048LL*512),256, 0, stream>>>(Ww2l,  Ww2l_bf,  2048*512, 2048,512, 512);
    k_convert_pad<<<nblk(2048LL*1024),256,0, stream>>>(Wgate, Wgate_bf, 2048*1024,2048,1024,1024);
    k_convert_pad<<<nblk(2048LL*512),256, 0, stream>>>(Whid,  Whid_bf,  2048*512, 2048,512, 512);
    k_convert_pad<<<nblk(2048LL*512),256, 0, stream>>>(Wboth, Wboth_bf, 2048*512, 2048,512, 512);
    k_convert_pad<<<nblk(512LL*512), 256, 0, stream>>>(Ww2lo, Ww2lo_bf, 512*512,  512, 512, 512);
    // word_embed[:-2] with 34 zero rows of padding (rows 11998..12031)
    k_convert_pad<<<nblk((long long)VP*512),256,0, stream>>>(wemb, wemb_bf, VP*512, 11998, 512, 512);

    // -------- precompute GEMMs --------
    gemm(stream, tags_bf, 0, 320, Wt1_bf, 0, 320, nullptr,0, nullptr,0,
         tmp2_f, 0, 2048, 128, 2048, 320, 1, 0, 0, 0, 0);
    gemm(stream, tags_bf, 0, 320, Wt2_bf, 0, 320, nullptr,0, nullptr,0,
         tmp4_f, 0, 2048, 128, 2048, 320, 1, 0, 0, 0, 0);
    gemm(stream, tags_bf, 0, 320, Wtg_bf, 0, 320, nullptr,0, nullptr,0,
         tg_f,   0, 2048, 128, 2048, 320, 1, 0, 0, 0, 0);
    gemm(stream, feats_bf,0, 2048, Wf2l_bf,0, 2048, nullptr,0, nullptr,0,
         tmp3_f, 0, 2048, 128, 2048, 2048, 1, 0, 0, 0, 0);
    gemm(stream, feats_bf,0, 2048, Wf2i_bf,0, 2048, nullptr,0, nullptr,0,
         we0_f,  0, 512, 128, 512, 2048, 1, 0, 0, 0, 0);

    // we (T,B,E) then tmp1 = we @ W_word2lstm^T -> bf16 (3200,2048)
    k_build_we<<<nblk(3200LL*512), 256, 0, stream>>>(we0_f, wemb, seq, we_bf);
    gemm(stream, we_bf, 0, 512, Ww2l_bf, 0, 512, nullptr,0, nullptr,0,
         tmp1_bf, 0, 2048, 3200, 2048, 512, 1, 1, 0, 0, 0);

    // second = tmp3*tmp4 ; inp_const = second @ W_gate[:,H:]^T + b_gate (z=4)
    k_prep_second<<<nblk(4LL*128*512), 256, 0, stream>>>(tmp3_f, tmp4_f, second_bf);
    gemm(stream, second_bf, 65536, 512, Wgate_bf + 512, 512LL*1024, 1024,
         nullptr, 0, bgate, 512, inpc_f, 65536, 512, 128, 512, 512, 4, 0, 0, 0, 0);

    // logit_w^T = word_embed_padded @ W_word2logit^T -> bf16 (VP,512); pad rows -> 0
    gemm(stream, wemb_bf, 0, 512, Ww2lo_bf, 0, 512, nullptr,0, nullptr,0,
         lwT_bf, 0, 512, VP, 512, 512, 1, 1, 0, 0, 0);

    hipMemsetAsync(h_bf, 0, (size_t)128 * 512 * 2, stream);
    hipMemsetAsync(c_f,  0, (size_t)128 * 512 * 4, stream);

    // -------- recurrence: T=25 steps --------
    for (int t = 0; t < 25; ++t) {
        k_prep_at<<<nblk(4LL*128*512), 256, 0, stream>>>(tmp1_bf, tmp2_f, a_bf, t);
        // inp = a @ W_gate[:, :H]^T + inp_const     (z=4)
        gemm(stream, a_bf, 65536, 512, Wgate_bf, 512LL*1024, 1024,
             inpc_f, 65536, nullptr, 0, inp_f, 65536, 512, 128, 512, 512, 4, 0, 0, 0, 0);
        // hid = h @ W_hid^T                         (z=4, A shared)
        gemm(stream, h_bf, 0, 512, Whid_bf, 512LL*512, 512,
             nullptr, 0, nullptr, 0, hid_f, 65536, 512, 128, 512, 512, 4, 0, 0, 0, 0);
        k_prep_hb<<<nblk(4LL*128*512), 256, 0, stream>>>(hid_f, tg_f, hb_bf);
        // pre = hb @ W_both^T + inp                 (z=4)
        gemm(stream, hb_bf, 65536, 512, Wboth_bf, 512LL*512, 512,
             inp_f, 65536, nullptr, 0, pre_f, 65536, 512, 128, 512, 512, 4, 0, 0, 0, 0);
        k_lstm<<<nblk(128LL*512), 256, 0, stream>>>(pre_f, c_f, h_bf, hs_bf, t);
    }

    // -------- logits = hs @ logit_w + bias, written in (b,t,v) order --------
    gemm(stream, hs_bf, 0, 512, lwT_bf, 0, 512, nullptr, 0, lbias, 0,
         d_out, 0, 11998, 3200, 11998, 512, 1, 0, /*remap=*/1, /*RB=*/128, /*RT=*/25);

    // in-place log_softmax over V-2 per (b,t) row
    k_logsoftmax<<<3200, 256, 0, stream>>>((float*)d_out, 11998);
}
